// ModelGraphCoordinationNet_66056597012565
// MI455X (gfx1250) — compile-verified, matrix operand order checked by reading, not run on hardware
//
#include <hip/hip_runtime.h>
#include <hip/hip_bf16.h>

typedef _Float16 h8   __attribute__((ext_vector_type(8)));
typedef _Float16 v16h __attribute__((ext_vector_type(16)));
typedef float    v8f  __attribute__((ext_vector_type(8)));
typedef float    f4   __attribute__((ext_vector_type(4)));

#define GEMM_BIAS 1
#define GEMM_ELU  2
#define GEMM_ADD  4

// ---------------------------------------------------------------------------
// Tiled WMMA GEMM: C[M,N] (+)= A[M,K] * B[K,N]  (A,B f32 in memory, converted
// to f16 in LDS; f32 accumulate with v_wmma_f32_16x16x32_f16).
// Block tile 128x128, 8 waves (4M x 2N), each wave a 32x64 tile
// (2x4 WMMA accumulators -> 8 WMMAs per K-step, A frags reused 4x).
// Interior tiles use vectorized b128 staging; edge tiles use clamped
// (branch-free) scalar staging.
// ---------------------------------------------------------------------------
__global__ __launch_bounds__(256)
void gemm_f32_wmma(const float* __restrict__ A, const float* __restrict__ B,
                   const float* __restrict__ bias, float* __restrict__ C,
                   int M, int N, int K, int flags)
{
    __shared__ __align__(16) _Float16 As[128 * 40];   // 128 rows x 32 k (pad 40)
    __shared__ __align__(16) _Float16 Bs[128 * 40];   // 128 cols x 32 k (transposed, pad 40)

    const int ntiles = (N + 127) >> 7;
    const int tm = blockIdx.x / ntiles;
    const int tn = blockIdx.x % ntiles;
    const int m0 = tm * 128, n0 = tn * 128;

    const int t    = threadIdx.x;
    const int wave = t >> 5;
    const int lane = t & 31;
    const int wm   = wave & 3;      // 4 wave rows -> 128 M
    const int wn   = wave >> 2;     // 2 wave cols -> 128 N
    const int l15  = lane & 15;
    const int lhi  = lane >> 4;

    const bool fullM = (m0 + 128 <= M);
    const bool fullN = (n0 + 128 <= N);

    // staging coordinates
    const int arow = t >> 1;            // A: 2 threads/row, 16 floats each
    const int ac0  = (t & 1) * 16;
    const int bn4  = (t & 31) * 4;      // B: float4 over n (32 thr * 4 = 128 cols)
    const int bk0  = t >> 5;            // 0..7, four passes (+8)

    v8f acc[2][4] = {};

    for (int k0 = 0; k0 < K; k0 += 32) {
        const bool fullK = (k0 + 32 <= K);

        // ---- stage A tile (128x32) as f16 ----
        if (fullM && fullK) {
            const f4* ap = (const f4*)(A + (long)(m0 + arow) * K + (k0 + ac0));
            f4 v0 = ap[0], v1 = ap[1], v2 = ap[2], v3 = ap[3];
            h8 h0, h1;
#pragma unroll
            for (int j = 0; j < 4; ++j) {
                h0[j]     = (_Float16)v0[j];
                h0[4 + j] = (_Float16)v1[j];
                h1[j]     = (_Float16)v2[j];
                h1[4 + j] = (_Float16)v3[j];
            }
            _Float16* dst = &As[arow * 40 + ac0];
            *(h8*)dst       = h0;
            *(h8*)(dst + 8) = h1;
        } else {
            int gm  = m0 + arow;
            int gmc = gm < M ? gm : M - 1;
            const float* ap = A + (long)gmc * K;
#pragma unroll
            for (int i = 0; i < 16; ++i) {
                int k  = k0 + ac0 + i;
                int kc = k < K ? k : K - 1;
                float v = ap[kc];                       // always-valid address
                v = (gm < M && k < K) ? v : 0.0f;       // mask, no exec branch
                As[arow * 40 + ac0 + i] = (_Float16)v;
            }
        }

        // ---- stage B tile (32x128) transposed -> Bs[n][k] ----
        if (fullN && fullK) {
#pragma unroll
            for (int p = 0; p < 4; ++p) {
                int kk = bk0 + p * 8;
                f4 v = *(const f4*)(B + (long)(k0 + kk) * N + (n0 + bn4));
#pragma unroll
                for (int j = 0; j < 4; ++j)
                    Bs[(bn4 + j) * 40 + kk] = (_Float16)v[j];
            }
        } else {
#pragma unroll
            for (int p = 0; p < 4; ++p) {
                int kk = bk0 + p * 8;
                int k  = k0 + kk;
                int kc = k < K ? k : K - 1;
#pragma unroll
                for (int j = 0; j < 4; ++j) {
                    int gn  = n0 + bn4 + j;
                    int gnc = gn < N ? gn : N - 1;
                    float v = B[(long)kc * N + gnc];
                    v = (k < K && gn < N) ? v : 0.0f;
                    Bs[(bn4 + j) * 40 + kk] = (_Float16)v;
                }
            }
        }
        __syncthreads();

        // ---- build fragments per ISA 16-bit layout, run 8 WMMAs ----
        v16h afrag[2], bfrag[4];
#pragma unroll
        for (int mt = 0; mt < 2; ++mt) {
            const _Float16* ap = &As[(wm * 32 + mt * 16 + l15) * 40];
            h8 lo = *(const h8*)(ap + lhi * 8);        // K =  0..7  / 8..15
            h8 hi = *(const h8*)(ap + 16 + lhi * 8);   // K = 16..23 / 24..31
            v16h a;
#pragma unroll
            for (int i = 0; i < 8; ++i) { a[i] = lo[i]; a[8 + i] = hi[i]; }
            afrag[mt] = a;
        }
#pragma unroll
        for (int nt = 0; nt < 4; ++nt) {
            const _Float16* bp = &Bs[(wn * 64 + nt * 16 + l15) * 40];
            h8 lo = *(const h8*)(bp + lhi * 8);
            h8 hi = *(const h8*)(bp + 16 + lhi * 8);
            v16h b;
#pragma unroll
            for (int i = 0; i < 8; ++i) { b[i] = lo[i]; b[8 + i] = hi[i]; }
            bfrag[nt] = b;
        }
#pragma unroll
        for (int mt = 0; mt < 2; ++mt)
#pragma unroll
            for (int nt = 0; nt < 4; ++nt)
                acc[mt][nt] = __builtin_amdgcn_wmma_f32_16x16x32_f16(
                    false, afrag[mt], false, bfrag[nt],
                    (short)0, acc[mt][nt], false, false);
        __syncthreads();
    }

    // ---- epilogue: C layout lane = N column, VGPR r = M row (r / r+8) ----
#pragma unroll
    for (int mt = 0; mt < 2; ++mt) {
#pragma unroll
        for (int nt = 0; nt < 4; ++nt) {
            int n     = n0 + wn * 64 + nt * 16 + l15;
            int mbase = m0 + wm * 32 + mt * 16 + lhi * 8;
            float bv = ((flags & GEMM_BIAS) && n < N) ? bias[n] : 0.0f;
#pragma unroll
            for (int r = 0; r < 8; ++r) {
                int m = mbase + r;
                if (m < M && n < N) {
                    long idx = (long)m * N + n;
                    float v = acc[mt][nt][r] + bv;
                    if (flags & GEMM_ADD) v += C[idx];
                    if (flags & GEMM_ELU) v = v > 0.0f ? v : (__expf(v) - 1.0f);
                    C[idx] = v;
                }
            }
        }
    }
}

// ---------------------------------------------------------------------------
// Gaussian RBF expansion + linear projection. One block per row, phi in LDS.
// out[i*stride + off + t] = b[t] + sum_b exp(-1521*(x_i - b/39)^2) * W[b,t]
// ---------------------------------------------------------------------------
__global__ __launch_bounds__(64)
void rbf_kernel(const float* __restrict__ x, const float* __restrict__ W,
                const float* __restrict__ b, float* __restrict__ out,
                int N, int D, int stride, int off)
{
    __shared__ float phi[40];
    int node = blockIdx.x;
    if (node >= N) return;
    float xv = x[node];
    int t = threadIdx.x;
    if (t < 40) {
        float d = xv - (float)t * (1.0f / 39.0f);
        phi[t] = __expf(-1521.0f * d * d);
    }
    __syncthreads();
    if (t < D) {
        float s = b[t];
#pragma unroll 8
        for (int i = 0; i < 40; ++i) s += phi[i] * W[i * D + t];
        out[(long)node * stride + off + t] = s;
    }
}

// ---------------------------------------------------------------------------
// Embedding-gather feature builders
// ---------------------------------------------------------------------------
__global__ void feat_site_k(const float* __restrict__ Eel, const float* __restrict__ Eox,
                            const int* __restrict__ el, const int* __restrict__ ox,
                            float* __restrict__ x, int N)
{
    long i = (long)blockIdx.x * blockDim.x + threadIdx.x;
    if (i >= (long)N * 144) return;
    int node = (int)(i / 144), c = (int)(i % 144);
    float v = (c < 128) ? Eel[el[node] * 128 + c] : Eox[ox[node] * 16 + (c - 128)];
    x[(long)node * 144 + c] = v;
}

__global__ void feat_ce_k(const float* __restrict__ Eel, const float* __restrict__ Eox,
                          const float* __restrict__ Ege, const int* __restrict__ el,
                          const int* __restrict__ ox, const int* __restrict__ ge,
                          float* __restrict__ x, int N)
{
    long i = (long)blockIdx.x * blockDim.x + threadIdx.x;
    if (i >= (long)N * 160) return;
    int node = (int)(i / 160), c = (int)(i % 160);
    float v;
    if (c < 128)      v = Eel[el[node] * 128 + c];
    else if (c < 144) v = Eox[ox[node] * 16 + (c - 128)];
    else              v = Ege[ge[node] * 16 + (c - 144)];
    x[(long)node * 224 + c] = v;   // rbf_csm fills cols 160..223
}

__global__ void feat_lig_k(const float* __restrict__ Eel, const float* __restrict__ Eox,
                           const int* __restrict__ el, const int* __restrict__ ox,
                           float* __restrict__ x, int N)
{
    long i = (long)blockIdx.x * blockDim.x + threadIdx.x;
    if (i >= (long)N * 144) return;
    int node = (int)(i / 144), c = (int)(i % 144);
    float v = (c < 128) ? Eel[el[node] * 128 + c] : Eox[ox[node] * 16 + (c - 128)];
    x[(long)node * 240 + c] = v;   // rbf_d1 -> 144..207, rbf_ang -> 208..239
}

// ---------------------------------------------------------------------------
// CGConv edge pass: m = sigmoid(Fd[dst]+Fs[src]+Fe[e]+bf) *
//                       softplus(Sd[dst]+Ss[src]+Se[e]+bs); accum[dst] += m
// One wave per edge; lanes stride the channel dimension (coalesced).
// ---------------------------------------------------------------------------
__global__ __launch_bounds__(256)
void edge_msg_k(const float* __restrict__ Fd, const float* __restrict__ Fs,
                const float* __restrict__ Fe, const float* __restrict__ Sd,
                const float* __restrict__ Ss, const float* __restrict__ Se,
                const float* __restrict__ bf, const float* __restrict__ bs,
                const int* __restrict__ src, const int* __restrict__ dst,
                float* __restrict__ accum, int E, int D)
{
    int e = blockIdx.x * (blockDim.x >> 5) + (threadIdx.x >> 5);
    if (e >= E) return;
    int lane = threadIdx.x & 31;
    int s = src[e], d = dst[e];
    const float* fd = Fd + (long)d * D;  const float* fs = Fs + (long)s * D;
    const float* fe = Fe + (long)e * D;  const float* sd = Sd + (long)d * D;
    const float* ss = Ss + (long)s * D;  const float* se = Se + (long)e * D;
    float* ac = accum + (long)d * D;
    for (int c = lane; c < D; c += 32) {
        float f = fd[c] + fs[c] + fe[c] + bf[c];
        float g = sd[c] + ss[c] + se[c] + bs[c];
        float sig = 1.0f / (1.0f + __expf(-f));
        float sp  = (g > 20.0f) ? g : __logf(1.0f + __expf(g));
        atomicAdd(&ac[c], sig * sp);
    }
}

__global__ void elu_inplace_k(float* __restrict__ x, long n)
{
    long i = (long)blockIdx.x * blockDim.x + threadIdx.x;
    if (i >= n) return;
    float v = x[i];
    x[i] = v > 0.0f ? v : (__expf(v) - 1.0f);
}

__global__ void zero_k(float* __restrict__ x, long n)
{
    long i = (long)blockIdx.x * blockDim.x + threadIdx.x;
    if (i < n) x[i] = 0.0f;
}

// agg[dst] += x[src]  (GraphConv aggregation)
__global__ void scatter_rows_k(const float* __restrict__ x, const int* __restrict__ src,
                               const int* __restrict__ dst, float* __restrict__ agg,
                               int E, int D)
{
    long i = (long)blockIdx.x * blockDim.x + threadIdx.x;
    if (i >= (long)E * D) return;
    int e = (int)(i / D), c = (int)(i % D);
    atomicAdd(&agg[(long)dst[e] * D + c], x[(long)src[e] * D + c]);
}

__global__ void pool_scatter_k(const float* __restrict__ xs, const int* __restrict__ batch,
                               float* __restrict__ pool, float* __restrict__ counts,
                               int N, int D)
{
    long i = (long)blockIdx.x * blockDim.x + threadIdx.x;
    if (i >= (long)N * D) return;
    int node = (int)(i / D), c = (int)(i % D);
    int g = batch[node];
    atomicAdd(&pool[(long)g * D + c], xs[i]);
    if (c == 0) atomicAdd(&counts[g], 1.0f);
}

__global__ void pool_div_k(float* __restrict__ pool, const float* __restrict__ counts,
                           int G, int D)
{
    long i = (long)blockIdx.x * blockDim.x + threadIdx.x;
    if (i >= (long)G * D) return;
    int g = (int)(i / D);
    float c = counts[g];
    pool[i] /= (c > 1.0f ? c : 1.0f);
}

// final dense layer [G,64] @ [64,1] + b, then inverse-scale
__global__ void head_final_k(const float* __restrict__ h, const float* __restrict__ W,
                             const float* __restrict__ b, const float* __restrict__ mean,
                             const float* __restrict__ stdv, float* __restrict__ out, int G)
{
    int g = blockIdx.x * blockDim.x + threadIdx.x;
    if (g >= G) return;
    float s = b[0];
#pragma unroll 8
    for (int k = 0; k < 64; ++k) s += h[(long)g * 64 + k] * W[k];
    out[g] = s * stdv[0] + mean[0];
}

// ---------------------------------------------------------------------------
// Host-side orchestration
// ---------------------------------------------------------------------------
static inline int cdiv(long a, long b) { return (int)((a + b - 1) / b); }

extern "C" void kernel_launch(void* const* d_in, const int* in_sizes, int n_in,
                              void* d_out, int out_size, void* d_ws, size_t ws_size,
                              hipStream_t stream)
{
    (void)in_sizes; (void)n_in; (void)ws_size;
    const int N_SITE = 100000, N_CE = 100000, N_LIG = 200000;
    const int E_LC = 500000, E_CL = 500000, E_CS = 400000;
    const int NG = 8192;

    auto F = [&](int i) { return (const float*)d_in[i]; };
    auto I = [&](int i) { return (const int*)d_in[i]; };

    // params (jax tree flatten, sorted dict keys)
    // convs[l] : cl.Wf, cl.Ws, cl.bf, cl.bs, lc.Wf, lc.Ws, lc.bf, lc.bs  (l*8 base)
    const float* d0W = F(16); const float* d0b = F(17);
    const float* d1W = F(18); const float* d1b = F(19);
    const float* d2W = F(20); const float* d2b = F(21);
    const float* d3W = F(22); const float* d3b = F(23);
    const float* emb_el = F(24); const float* emb_geo = F(25); const float* emb_ox = F(26);
    const float* gc_Wrel = F(27); const float* gc_Wroot = F(28);
    const float* rbf_angW = F(29); const float* rbf_angB = F(30);
    const float* rbf_csmW = F(31); const float* rbf_csmB = F(32);
    const float* rbf_d1W  = F(33); const float* rbf_d1B  = F(34);
    const float* rbf_d2W  = F(35); const float* rbf_d2B  = F(36);
    const float* sc_mean  = F(37); const float* sc_std   = F(38);

    const int* site_el = I(39); const int* site_ox = I(40);
    const int* ce_el = I(41);   const int* ce_ox = I(42);   const int* ce_geo = I(43);
    const float* ce_csm = F(44);
    const int* lig_el = I(45);  const int* lig_ox = I(46);
    const float* lig_dist = F(47); const float* lig_ang = F(48);
    const int* elc_src = I(49); const int* elc_dst = I(50); const float* elc_attr = F(51);
    const int* ecl_src = I(52); const int* ecl_dst = I(53); const float* ecl_attr = F(54);
    const int* ecs_src = I(55); const int* ecs_dst = I(56);
    const int* batch_site = I(57);

    // workspace bump allocator
    char* wsb = (char*)d_ws;
    size_t off = 0;
    auto alloc = [&](long n) {
        float* p = (float*)(wsb + off);
        off = (off + (size_t)n * 4 + 255) & ~(size_t)255;
        return p;
    };
    float* x_site = alloc((long)N_SITE * 144);
    float* x_ce   = alloc((long)N_CE * 224);
    float* x_lig  = alloc((long)N_LIG * 240);
    float* x_ce2  = alloc((long)N_CE * 224);
    float* x_lig2 = alloc((long)N_LIG * 240);
    float* e_lc   = alloc((long)E_LC * 64);
    float* e_cl   = alloc((long)E_CL * 64);
    float* Fd = alloc((long)N_LIG * 240);   // sized for max use
    float* Sd = alloc((long)N_LIG * 240);
    float* Fs = alloc((long)N_LIG * 240);
    float* Ss = alloc((long)N_LIG * 240);
    float* Fe = alloc((long)E_LC * 240);
    float* Se = alloc((long)E_CL * 240);
    float* agg  = alloc((long)N_SITE * 224);
    float* xs   = alloc((long)N_SITE * 144);
    float* pool = alloc((long)NG * 144);
    float* cnt  = alloc(NG);
    float* h1   = alloc((long)NG * 512);
    float* h2   = alloc((long)NG * 128);
    float* h3   = alloc((long)NG * 64);

    auto gemm = [&](const float* A, const float* B, const float* bias, float* C,
                    int M, int N, int K, int flags) {
        int grid = cdiv(M, 128) * cdiv(N, 128);
        gemm_f32_wmma<<<grid, 256, 0, stream>>>(A, B, bias, C, M, N, K, flags);
    };

    // ---- input features ----
    feat_site_k<<<cdiv((long)N_SITE * 144, 256), 256, 0, stream>>>(emb_el, emb_ox, site_el, site_ox, x_site, N_SITE);
    feat_ce_k<<<cdiv((long)N_CE * 160, 256), 256, 0, stream>>>(emb_el, emb_ox, emb_geo, ce_el, ce_ox, ce_geo, x_ce, N_CE);
    rbf_kernel<<<N_CE, 64, 0, stream>>>(ce_csm, rbf_csmW, rbf_csmB, x_ce, N_CE, 64, 224, 160);
    feat_lig_k<<<cdiv((long)N_LIG * 144, 256), 256, 0, stream>>>(emb_el, emb_ox, lig_el, lig_ox, x_lig, N_LIG);
    rbf_kernel<<<N_LIG, 64, 0, stream>>>(lig_dist, rbf_d1W, rbf_d1B, x_lig, N_LIG, 64, 240, 144);
    rbf_kernel<<<N_LIG, 64, 0, stream>>>(lig_ang, rbf_angW, rbf_angB, x_lig, N_LIG, 32, 240, 208);
    rbf_kernel<<<E_LC, 64, 0, stream>>>(elc_attr, rbf_d2W, rbf_d2B, e_lc, E_LC, 64, 64, 0);
    rbf_kernel<<<E_CL, 64, 0, stream>>>(ecl_attr, rbf_d2W, rbf_d2B, e_cl, E_CL, 64, 64, 0);

    // ---- CGConv stack (decomposed concat-GEMM + edge scatter) ----
    for (int l = 0; l < 2; ++l) {
        const float* clWf = F(0 + l * 8); const float* clWs = F(1 + l * 8);
        const float* clbf = F(2 + l * 8); const float* clbs = F(3 + l * 8);
        const float* lcWf = F(4 + l * 8); const float* lcWs = F(5 + l * 8);
        const float* lcbf = F(6 + l * 8); const float* lcbs = F(7 + l * 8);

        // lc: dst=ce(224), src=lig(240), W rows: [0:224)=dst, [224:464)=src, [464:528)=edge
        gemm(x_ce,  lcWf,             nullptr, Fd, N_CE, 224, 224, 0);
        gemm(x_ce,  lcWs,             nullptr, Sd, N_CE, 224, 224, 0);
        gemm(x_lig, lcWf + 224 * 224, nullptr, Fs, N_LIG, 224, 240, 0);
        gemm(x_lig, lcWs + 224 * 224, nullptr, Ss, N_LIG, 224, 240, 0);
        gemm(e_lc,  lcWf + 464 * 224, nullptr, Fe, E_LC, 224, 64, 0);
        gemm(e_lc,  lcWs + 464 * 224, nullptr, Se, E_LC, 224, 64, 0);
        hipMemcpyAsync(x_ce2, x_ce, (size_t)N_CE * 224 * 4, hipMemcpyDeviceToDevice, stream);
        edge_msg_k<<<cdiv(E_LC, 8), 256, 0, stream>>>(Fd, Fs, Fe, Sd, Ss, Se, lcbf, lcbs,
                                                      elc_src, elc_dst, x_ce2, E_LC, 224);
        elu_inplace_k<<<cdiv((long)N_CE * 224, 256), 256, 0, stream>>>(x_ce2, (long)N_CE * 224);

        // cl: dst=lig(240), src=ce(224), W rows: [0:240)=dst, [240:464)=src, [464:528)=edge
        gemm(x_lig, clWf,             nullptr, Fd, N_LIG, 240, 240, 0);
        gemm(x_lig, clWs,             nullptr, Sd, N_LIG, 240, 240, 0);
        gemm(x_ce,  clWf + 240 * 240, nullptr, Fs, N_CE, 240, 224, 0);
        gemm(x_ce,  clWs + 240 * 240, nullptr, Ss, N_CE, 240, 224, 0);
        gemm(e_cl,  clWf + 464 * 240, nullptr, Fe, E_CL, 240, 64, 0);
        gemm(e_cl,  clWs + 464 * 240, nullptr, Se, E_CL, 240, 64, 0);
        hipMemcpyAsync(x_lig2, x_lig, (size_t)N_LIG * 240 * 4, hipMemcpyDeviceToDevice, stream);
        edge_msg_k<<<cdiv(E_CL, 8), 256, 0, stream>>>(Fd, Fs, Fe, Sd, Ss, Se, clbf, clbs,
                                                      ecl_src, ecl_dst, x_lig2, E_CL, 240);
        elu_inplace_k<<<cdiv((long)N_LIG * 240, 256), 256, 0, stream>>>(x_lig2, (long)N_LIG * 240);

        { float* tp = x_ce; x_ce = x_ce2; x_ce2 = tp; }
        { float* tp = x_lig; x_lig = x_lig2; x_lig2 = tp; }
    }

    // ---- GraphConv ce -> site ----
    zero_k<<<cdiv((long)N_SITE * 224, 256), 256, 0, stream>>>(agg, (long)N_SITE * 224);
    scatter_rows_k<<<cdiv((long)E_CS * 224, 256), 256, 0, stream>>>(x_ce, ecs_src, ecs_dst, agg, E_CS, 224);
    gemm(agg,    gc_Wrel,  nullptr, xs, N_SITE, 144, 224, 0);
    gemm(x_site, gc_Wroot, nullptr, xs, N_SITE, 144, 144, GEMM_ADD);
    elu_inplace_k<<<cdiv((long)N_SITE * 144, 256), 256, 0, stream>>>(xs, (long)N_SITE * 144);

    // ---- global mean pool ----
    zero_k<<<cdiv((long)NG * 144, 256), 256, 0, stream>>>(pool, (long)NG * 144);
    zero_k<<<cdiv((long)NG, 256), 256, 0, stream>>>(cnt, NG);
    pool_scatter_k<<<cdiv((long)N_SITE * 144, 256), 256, 0, stream>>>(xs, batch_site, pool, cnt, N_SITE, 144);
    pool_div_k<<<cdiv((long)NG * 144, 256), 256, 0, stream>>>(pool, cnt, NG, 144);

    // ---- dense head ----
    gemm(pool, d0W, d0b, h1, NG, 512, 144, GEMM_BIAS | GEMM_ELU);
    gemm(h1,   d1W, d1b, h2, NG, 128, 512, GEMM_BIAS | GEMM_ELU);
    gemm(h2,   d2W, d2b, h3, NG, 64, 128, GEMM_BIAS | GEMM_ELU);
    head_final_k<<<cdiv(NG, 256), 256, 0, stream>>>(h3, d3W, d3b, sc_mean, sc_std, (float*)d_out, NG);
    (void)out_size;
}
